// InfoNCECauchyAUCCL_80238579024621
// MI455X (gfx1250) — matrix operand order repaired and checked by reading, not compiled, not run on hardware
//
#include <hip/hip_runtime.h>

// ---------------------------------------------------------------------------
// InfoNCE/Cauchy loss on MI455X (gfx1250, wave32, WMMA).
// Dominant cost: two 8192x8192x128 Gram matrices -> bf16 hi/lo split WMMA.
// ---------------------------------------------------------------------------

typedef __attribute__((ext_vector_type(16))) __bf16 v16bf;
typedef __attribute__((ext_vector_type(8)))  float  v8f;

union Frag {
    uint4 q[2];
    v16bf v;
};

#define DDIM   128
#define TILE   128
#define NTILE  64        // 8192 / 128
#define STRIDE 136       // 128 + 8 bf16 pad: rows offset by 4 banks -> conflict-free b128 reads
#define NPAIRS 2080      // 64*65/2 upper-triangular tiles
#define INV_B  (1.0f / 8192.0f)

__device__ inline unsigned short f2bf(float f) {
    union { float f; unsigned u; } x; x.f = f;
    unsigned r = x.u + 0x7FFFu + ((x.u >> 16) & 1u);   // round-to-nearest-even
    return (unsigned short)(r >> 16);
}
__device__ inline float bf2f(unsigned short h) {
    union { unsigned u; float f; } x; x.u = ((unsigned)h) << 16;
    return x.f;
}
__device__ inline unsigned pack2(unsigned short a, unsigned short b) {
    return (unsigned)a | ((unsigned)b << 16);
}

// ---------------------------------------------------------------------------
// Kernel 0: zero the scalar output.
// ---------------------------------------------------------------------------
__global__ void zero_kernel(float* out) { out[0] = 0.0f; }

// ---------------------------------------------------------------------------
// Kernel 1: row norms for all 16384 rows + positive-pair loss.
// One wave32 per row; lane loads float4 (32 lanes * 4 = 128 cols).
// ---------------------------------------------------------------------------
__global__ void __launch_bounds__(256)
norms_pos_kernel(const float* __restrict__ X, float* __restrict__ norms,
                 float* __restrict__ out) {
    const int lane = threadIdx.x & 31;
    const int w    = threadIdx.x >> 5;
    const int row  = blockIdx.x * 8 + w;

    const float4* p = (const float4*)(X + (size_t)row * DDIM);
    float4 v = p[lane];
    float s  = v.x * v.x + v.y * v.y + v.z * v.z + v.w * v.w;

    float d2 = 0.0f;
    if (row < 8192) {
        const float4* q = (const float4*)(X + (size_t)(row + 8192) * DDIM);
        float4 u  = q[lane];
        float dx = v.x - u.x, dy = v.y - u.y, dz = v.z - u.z, dw = v.w - u.w;
        d2 = dx * dx + dy * dy + dz * dz + dw * dw;
    }
    #pragma unroll
    for (int off = 16; off > 0; off >>= 1) {
        s  += __shfl_xor(s,  off, 32);
        d2 += __shfl_xor(d2, off, 32);
    }
    if (lane == 0) {
        norms[row] = s;
        if (row < 8192) {
            float ps = 1.0f / (d2 + 1.0f);                  // TEMP^2 = 1
            float c  = ((ps - 10.0f) * (ps - 10.0f) - ps);  // -2*ALPHA*ps = -ps
            atomicAdd(out, c * INV_B);
        }
    }
}

// ---------------------------------------------------------------------------
// Kernel 2: one workgroup per upper-triangular 128x128 tile of sim(x,x).
// blockIdx.y selects matrix a (0) or b (1).  bf16 hi/lo split WMMA.
// ---------------------------------------------------------------------------
__global__ void __launch_bounds__(256)
sim_tile_kernel(const float* __restrict__ Xall, const float* __restrict__ Nall,
                float* __restrict__ out) {
    extern __shared__ unsigned short smem[];
    unsigned short* Ihi = smem;
    unsigned short* Ilo = Ihi + 128 * STRIDE;
    unsigned short* Jhi = Ilo + 128 * STRIDE;
    unsigned short* Jlo = Jhi + 128 * STRIDE;
    float* nI = (float*)(Jlo + 128 * STRIDE);
    float* nJ = nI + 128;

    const float* X    = Xall + (size_t)blockIdx.y * 8192 * DDIM;
    const float* NORM = Nall + (size_t)blockIdx.y * 8192;

    // decode upper-triangular tile pair (ti <= tj)
    int t = blockIdx.x, ti = 0;
    while (t >= NTILE - ti) { t -= NTILE - ti; ++ti; }
    const int tj   = ti + t;
    const bool diag = (ti == tj);
    const int i0 = ti * TILE, j0 = tj * TILE;

    const int tid = threadIdx.x;

    // ---- stage: global fp32 -> LDS bf16 hi/lo (split) ----
    {
        const int r    = tid >> 1;
        const int half = (tid & 1) * 64;
        #pragma unroll
        for (int blk = 0; blk < 2; ++blk) {
            const float* src = X + (size_t)((blk ? j0 : i0) + r) * DDIM + half;
            unsigned short* dh = (blk ? Jhi : Ihi) + r * STRIDE + half;
            unsigned short* dl = (blk ? Jlo : Ilo) + r * STRIDE + half;
            #pragma unroll
            for (int c = 0; c < 64; c += 8) {
                float4 f0 = *(const float4*)(src + c);
                float4 f1 = *(const float4*)(src + c + 4);
                float fs[8] = {f0.x, f0.y, f0.z, f0.w, f1.x, f1.y, f1.z, f1.w};
                unsigned short hs[8], ls[8];
                #pragma unroll
                for (int q = 0; q < 8; ++q) {
                    hs[q] = f2bf(fs[q]);
                    ls[q] = f2bf(fs[q] - bf2f(hs[q]));
                }
                *(uint4*)(dh + c) = make_uint4(pack2(hs[0], hs[1]), pack2(hs[2], hs[3]),
                                               pack2(hs[4], hs[5]), pack2(hs[6], hs[7]));
                *(uint4*)(dl + c) = make_uint4(pack2(ls[0], ls[1]), pack2(ls[2], ls[3]),
                                               pack2(ls[4], ls[5]), pack2(ls[6], ls[7]));
            }
        }
        if (tid < 128) nI[tid]       = NORM[i0 + tid];
        else           nJ[tid - 128] = NORM[j0 + (tid - 128)];
    }
    __syncthreads();

    // ---- compute: each wave owns 64 rows x 32 cols (4x2 subtiles of 16x16) ----
    const int lane = tid & 31;
    const int w    = tid >> 5;
    const int wr   = w >> 2;       // 0..1 -> row base wr*64
    const int wc   = w & 3;        // 0..3 -> col base wc*32
    const int lm   = lane & 15;
    const int lh   = lane >> 4;

    v8f acc[4][2];
    #pragma unroll
    for (int r = 0; r < 4; ++r)
        #pragma unroll
        for (int c = 0; c < 2; ++c)
            acc[r][c] = v8f{0.f, 0.f, 0.f, 0.f, 0.f, 0.f, 0.f, 0.f};

    #pragma unroll
    for (int k0 = 0; k0 < 128; k0 += 32) {
        #pragma unroll
        for (int r = 0; r < 4; ++r) {
            // A fragment (16x32 bf16, ISA layout): lane<16 rows m=lm K={k0..k0+7, k0+16..23},
            // lane>=16 K offset +8 -> two b128 reads at +0 and +32 bytes.
            const int arow = wr * 64 + r * 16 + lm;
            const uint4* pAh = (const uint4*)(Ihi + arow * STRIDE + k0 + lh * 8);
            const uint4* pAl = (const uint4*)(Ilo + arow * STRIDE + k0 + lh * 8);
            Frag ah, al;
            ah.q[0] = pAh[0]; ah.q[1] = pAh[2];
            al.q[0] = pAl[0]; al.q[1] = pAl[2];
            #pragma unroll
            for (int c = 0; c < 2; ++c) {
                // B fragment (32x16 bf16): lane<16 col n=lm K={k0..k0+7, k0+16..23}, lane>=16 K offset +8
                // -> two b128 reads at +0 and +32 bytes (same K order as A).
                const int brow = wc * 32 + c * 16 + lm;
                const uint4* pBh = (const uint4*)(Jhi + brow * STRIDE + k0 + lh * 8);
                const uint4* pBl = (const uint4*)(Jlo + brow * STRIDE + k0 + lh * 8);
                Frag bh, bl;
                bh.q[0] = pBh[0]; bh.q[1] = pBh[2];
                bl.q[0] = pBl[0]; bl.q[1] = pBl[2];
                // dot ~= hi*hi + hi*lo + lo*hi  (lo*lo dropped; ~2^-16 rel err)
                acc[r][c] = __builtin_amdgcn_wmma_f32_16x16x32_bf16(
                    false, ah.v, false, bh.v, (short)0, acc[r][c], false, false);
                acc[r][c] = __builtin_amdgcn_wmma_f32_16x16x32_bf16(
                    false, ah.v, false, bl.v, (short)0, acc[r][c], false, false);
                acc[r][c] = __builtin_amdgcn_wmma_f32_16x16x32_bf16(
                    false, al.v, false, bh.v, (short)0, acc[r][c], false, false);
            }
        }
    }

    // ---- epilogue: sim, diagonal zero, fused neg-loss + gamma regularizer ----
    float lsum = 0.0f;
    #pragma unroll
    for (int r = 0; r < 4; ++r) {
        #pragma unroll
        for (int c = 0; c < 2; ++c) {
            const int lj  = wc * 32 + c * 16 + lm;
            const float xxj = nJ[lj];
            #pragma unroll
            for (int v = 0; v < 8; ++v) {
                const int li  = wr * 64 + r * 16 + (lh << 3) + v;  // C/D layout: M = v + 8*lh
                const float dot = acc[r][c][v];
                float dd  = fmaxf(nI[li] + xxj - 2.0f * dot, 0.0f);
                float sim = 1.0f / (dd + 1.0f);
                if (diag && (li == lj)) sim = 0.0f;
                // (1/B)*exp(BETA*sim)*sim + GAMMA*sim, BETA=GAMMA=0.5
                lsum += sim * (__expf(0.5f * sim) * INV_B + 0.5f);
            }
        }
    }
    if (!diag) lsum *= 2.0f;   // mirror tile (tj, ti)

    #pragma unroll
    for (int off = 16; off > 0; off >>= 1) lsum += __shfl_xor(lsum, off, 32);
    if (lane == 0) atomicAdd(out, lsum);
}

// ---------------------------------------------------------------------------
extern "C" void kernel_launch(void* const* d_in, const int* in_sizes, int n_in,
                              void* d_out, int out_size, void* d_ws, size_t ws_size,
                              hipStream_t stream) {
    (void)in_sizes; (void)n_in; (void)out_size; (void)ws_size;
    const float* X   = (const float*)d_in[0];
    float* out       = (float*)d_out;
    float* norms     = (float*)d_ws;          // 16384 floats = 64 KB scratch

    zero_kernel<<<1, 1, 0, stream>>>(out);
    norms_pos_kernel<<<16384 / 8, 256, 0, stream>>>(X, norms, out);

    dim3 grid(NPAIRS, 2);                     // upper-tri tiles x {a, b}
    size_t smem = (size_t)4 * 128 * STRIDE * sizeof(unsigned short)
                + 256 * sizeof(float);        // ~140 KB (320 KB WGP LDS available)
    sim_tile_kernel<<<grid, 256, smem, stream>>>(X, norms, out);
}
